// MultiHeadAttention_20925080666463
// MI455X (gfx1250) — matrix operand order repaired and checked by reading, not compiled, not run on hardware
//
#include <hip/hip_runtime.h>
#include <hip/hip_bf16.h>

// ---------------- CDNA5 WMMA types ----------------
typedef __attribute__((ext_vector_type(16))) __bf16 v16bf;
typedef __attribute__((ext_vector_type(8)))  float  v8f;

__device__ __forceinline__ v8f wmma_bf16(v16bf a, v16bf b, v8f c) {
    // v_wmma_f32_16x16x32_bf16  (neg_a, A, neg_b, B, c_mod, C, reuse_a, reuse_b)
    return __builtin_amdgcn_wmma_f32_16x16x32_bf16(false, a, false, b, (short)0, c,
                                                   false, false);
}

union FragU {
    uint4  q[2];
    v16bf  v;
    __bf16 e[16];
};

// CDNA5 async memory->LDS copy (16B per lane), tracked by ASYNCcnt.
// vdst = LDS byte address (low 32 bits of generic shared pointer, per aperture rule),
// vaddr = 64-bit global address, GV mode (saddr = off).
__device__ __forceinline__ void async_load_lds_b128(const void* gptr, void* lptr) {
    unsigned lds = (unsigned)(size_t)lptr;
    asm volatile("global_load_async_to_lds_b128 %0, %1, off"
                 :: "v"(lds), "v"(gptr) : "memory");
}
__device__ __forceinline__ void wait_asynccnt0() {
    asm volatile("s_wait_asynccnt 0x0" ::: "memory");
}

// Problem constants
#define BATCH 8
#define SEQ   1024
#define DMODEL 512
#define HEADS 8
#define HK    64          // per-head q/k dim
#define HD    512         // per-head value dim (== DMODEL)

// ---------------- f32 -> bf16 convert ----------------
__global__ void __launch_bounds__(256) cvt_bf16_kernel(const float* __restrict__ in,
                                                       __bf16* __restrict__ out, int n) {
    int i = blockIdx.x * 256 + threadIdx.x;
    int stride = gridDim.x * 256;
    for (; i < n; i += stride) out[i] = (__bf16)in[i];
}

// ---------------- Generic batched bf16 WMMA GEMM ----------------
// C[M,N] = A[M,Kd] @ B[Kd,N]; A,B bf16 row-major (lda/ldb), C bf16 or f32 (ldc).
// blockDim 256 (8 waves); tile 128x128, K-step 64 (two 32-wide WMMA sub-steps).
// Each wave: 32(M) x 64(N) -> 16 v_wmma per K-tile.
// A tile staged via GLOBAL_LOAD_ASYNC_TO_LDS_B128 (ASYNCcnt); B tile staged
// transposed through VGPRs (async copy cannot transpose).
__global__ void __launch_bounds__(256) gemm_bf16_kernel(
    const __bf16* __restrict__ A, const __bf16* __restrict__ B, void* __restrict__ C,
    int M, int N, int Kd, int lda, int ldb, int ldc,
    long long sAz, long long sBz, long long sCz, int storeBf16) {

    __shared__ __bf16 sA[128][72];   // [m][k]  row stride 144B
    __shared__ __bf16 sB[128][72];   // [n][k]  (B tile stored transposed)

    const long long z = blockIdx.z;
    A += z * sAz;
    B += z * sBz;
    __bf16* Cb = (__bf16*)C + z * sCz;
    float*  Cf = (float*)C  + z * sCz;

    const int tid  = threadIdx.x;
    const int wave = tid >> 5;
    const int lane = tid & 31;
    const int wm   = wave >> 1;     // 0..3  -> M sub-tile of 32
    const int wn   = wave & 1;      // 0..1  -> N sub-tile of 64
    const int r16  = lane & 15;
    const int kh   = lane >> 4;     // 0/1 half selector

    const size_t row0 = (size_t)blockIdx.y * 128;
    const size_t col0 = (size_t)blockIdx.x * 128;

    v8f acc[2][4] = {};

    for (int k0 = 0; k0 < Kd; k0 += 64) {
        // ---- stage A tile 128x64 via async copy to LDS ----
        #pragma unroll
        for (int i = 0; i < 4; ++i) {
            int c   = tid * 4 + i;          // 0..1023 chunks of 8 bf16
            int row = c >> 3;
            int cc  = c & 7;
            async_load_lds_b128(A + (row0 + row) * (size_t)lda + k0 + cc * 8,
                                &sA[row][cc * 8]);
        }
        // ---- stage B tile 64x128 transposed -> sB[n][k] ----
        #pragma unroll
        for (int i = 0; i < 4; ++i) {
            int c    = tid * 4 + i;         // 0..1023 chunks of 8 bf16 along n
            int krow = c >> 4;
            int nc   = c & 15;
            const uint4 v = *(const uint4*)(B + (size_t)(k0 + krow) * ldb + col0 + nc * 8);
            FragU u; u.q[0] = v;
            #pragma unroll
            for (int j = 0; j < 8; ++j) sB[nc * 8 + j][krow] = u.e[j];
            // speculative prefetch of the next K tile (silently dropped if OOB)
            __builtin_prefetch(B + (size_t)(k0 + 64 + krow) * ldb + col0 + nc * 8, 0, 1);
        }
        wait_asynccnt0();
        __syncthreads();

        // ---- two 32-wide WMMA sub-steps over the staged 64-K tile ----
        #pragma unroll
        for (int kk = 0; kk < 64; kk += 32) {
            // A fragments (16x32): lane<16 K0-7|16-23, lane>=16 K8-15|24-31
            v16bf afrag[2];
            #pragma unroll
            for (int mi = 0; mi < 2; ++mi) {
                int row = wm * 32 + mi * 16 + r16;
                FragU u;
                u.q[0] = *(const uint4*)&sA[row][kk + kh * 8];
                u.q[1] = *(const uint4*)&sA[row][kk + 16 + kh * 8];
                afrag[mi] = u.v;
            }
            // B fragments (32x16): lane<16 K0-7|16-23, lane>=16 K8-15|24-31, col=n
            #pragma unroll
            for (int ni = 0; ni < 4; ++ni) {
                int ncol = wn * 64 + ni * 16 + r16;
                FragU u;
                u.q[0] = *(const uint4*)&sB[ncol][kk + kh * 8];
                u.q[1] = *(const uint4*)&sB[ncol][kk + 16 + kh * 8];
                v16bf bfrag = u.v;
                acc[0][ni] = wmma_bf16(afrag[0], bfrag, acc[0][ni]);
                acc[1][ni] = wmma_bf16(afrag[1], bfrag, acc[1][ni]);
            }
        }
        __syncthreads();
    }

    // ---- store: C tile row = r + 8*kh (lanes 16-31 hold M=r+8), col = r16 ----
    #pragma unroll
    for (int mi = 0; mi < 2; ++mi)
        #pragma unroll
        for (int ni = 0; ni < 4; ++ni)
            #pragma unroll
            for (int r = 0; r < 8; ++r) {
                size_t grow = row0 + wm * 32 + mi * 16 + r + 8 * kh;
                size_t gcol = col0 + wn * 64 + ni * 16 + r16;
                float val = acc[mi][ni][r];
                if (storeBf16) Cb[grow * (size_t)ldc + gcol] = (__bf16)val;
                else           Cf[grow * (size_t)ldc + gcol] = val;
            }
}

// ---------------- attention scores + online softmax -> att (bf16) ----------------
// One head per launch (h param); grid = (SEQ/128, BATCH); block 256 (8 waves).
// Wave owns 16 query rows; two passes: (1) streaming max/sum, (2) normalized write.
__global__ void __launch_bounds__(256) attn_softmax_kernel(
    const __bf16* __restrict__ Qb,   // [B,S,H,HK]
    const __bf16* __restrict__ Kb,   // [B,S,H,HK]
    __bf16* __restrict__ att,        // [B,Sq,Sk] for this head
    int h) {

    const int b    = blockIdx.y;
    const int wave = threadIdx.x >> 5;
    const int lane = threadIdx.x & 31;
    const int r16  = lane & 15;
    const int kh   = lane >> 4;
    const int qbase = blockIdx.x * 128 + wave * 16;

    // Q fragments (rows qbase..+15, k 0..63 -> two 16x32 frags), loaded once.
    const __bf16* qp = Qb + (((size_t)b * SEQ + qbase + r16) * HEADS + h) * HK;
    FragU uq0, uq1;
    uq0.q[0] = *(const uint4*)(qp + kh * 8);
    uq0.q[1] = *(const uint4*)(qp + 16 + kh * 8);
    uq1.q[0] = *(const uint4*)(qp + 32 + kh * 8);
    uq1.q[1] = *(const uint4*)(qp + 48 + kh * 8);
    const v16bf q0 = uq0.v, q1 = uq1.v;

    float m[8], l[8];
    #pragma unroll
    for (int r = 0; r < 8; ++r) { m[r] = -1e30f; l[r] = 0.0f; }

    // ---------- pass 1: streaming softmax stats ----------
    for (int j0 = 0; j0 < SEQ; j0 += 16) {
        const __bf16* kp = Kb + (((size_t)b * SEQ + j0 + r16) * HEADS + h) * HK;
        FragU ub0, ub1;
        ub0.q[0] = *(const uint4*)(kp + kh * 8);
        ub0.q[1] = *(const uint4*)(kp + 16 + kh * 8);
        ub1.q[0] = *(const uint4*)(kp + 32 + kh * 8);
        ub1.q[1] = *(const uint4*)(kp + 48 + kh * 8);
        v8f acc = {};
        acc = wmma_bf16(q0, ub0.v, acc);
        acc = wmma_bf16(q1, ub1.v, acc);
        #pragma unroll
        for (int r = 0; r < 8; ++r) {
            float s = acc[r] * 0.125f;              // 1/sqrt(64)
            if (s == 0.0f) s = -1e9f;               // value-based mask
            float t = s;
            t = fmaxf(t, __shfl_xor(t, 1, 32));
            t = fmaxf(t, __shfl_xor(t, 2, 32));
            t = fmaxf(t, __shfl_xor(t, 4, 32));
            t = fmaxf(t, __shfl_xor(t, 8, 32));
            float mnew = fmaxf(m[r], t);
            float p = __expf(s - mnew);
            float ps = p;
            ps += __shfl_xor(ps, 1, 32);
            ps += __shfl_xor(ps, 2, 32);
            ps += __shfl_xor(ps, 4, 32);
            ps += __shfl_xor(ps, 8, 32);
            l[r] = l[r] * __expf(m[r] - mnew) + ps;
            m[r] = mnew;
        }
    }
    float rl[8];
    #pragma unroll
    for (int r = 0; r < 8; ++r) rl[r] = 1.0f / l[r];

    // ---------- pass 2: recompute, normalize, emit bf16 ----------
    for (int j0 = 0; j0 < SEQ; j0 += 16) {
        const __bf16* kp = Kb + (((size_t)b * SEQ + j0 + r16) * HEADS + h) * HK;
        FragU ub0, ub1;
        ub0.q[0] = *(const uint4*)(kp + kh * 8);
        ub0.q[1] = *(const uint4*)(kp + 16 + kh * 8);
        ub1.q[0] = *(const uint4*)(kp + 32 + kh * 8);
        ub1.q[1] = *(const uint4*)(kp + 48 + kh * 8);
        v8f acc = {};
        acc = wmma_bf16(q0, ub0.v, acc);
        acc = wmma_bf16(q1, ub1.v, acc);
        #pragma unroll
        for (int r = 0; r < 8; ++r) {
            float s = acc[r] * 0.125f;
            if (s == 0.0f) s = -1e9f;
            float p = __expf(s - m[r]) * rl[r];
            size_t grow = (size_t)(qbase + r + 8 * kh);
            att[(size_t)b * SEQ * SEQ + grow * SEQ + j0 + r16] = (__bf16)p;
        }
    }
}

// ---------------- bias + residual + LayerNorm epilogue ----------------
__global__ void __launch_bounds__(256) resid_ln_kernel(
    const float* __restrict__ proj, const float* __restrict__ x,
    const float* __restrict__ bo, const float* __restrict__ gamma,
    const float* __restrict__ beta, float* __restrict__ out) {
    const int row = blockIdx.x;
    const int tid = threadIdx.x;
    __shared__ float s1[256], s2[256];
    const size_t base = (size_t)row * DMODEL;
    float y0 = proj[base + tid]       + bo[tid]       + x[base + tid];
    float y1 = proj[base + 256 + tid] + bo[256 + tid] + x[base + 256 + tid];
    s1[tid] = y0 + y1;
    s2[tid] = y0 * y0 + y1 * y1;
    __syncthreads();
    for (int off = 128; off > 0; off >>= 1) {
        if (tid < off) { s1[tid] += s1[tid + off]; s2[tid] += s2[tid + off]; }
        __syncthreads();
    }
    const float mu  = s1[0] * (1.0f / DMODEL);
    const float var = s2[0] * (1.0f / DMODEL) - mu * mu;
    const float rstd = rsqrtf(var + 1e-6f);
    out[base + tid]       = (y0 - mu) * rstd * gamma[tid]       + beta[tid];
    out[base + 256 + tid] = (y1 - mu) * rstd * gamma[256 + tid] + beta[256 + tid];
}

// ---------------- host-side orchestration ----------------
extern "C" void kernel_launch(void* const* d_in, const int* in_sizes, int n_in,
                              void* d_out, int out_size, void* d_ws, size_t ws_size,
                              hipStream_t stream) {
    (void)in_sizes; (void)n_in; (void)out_size; (void)ws_size;
    const float* x     = (const float*)d_in[0];   // [B,S,D]
    const float* Wq    = (const float*)d_in[1];   // [D, H*K]
    const float* Wk    = (const float*)d_in[2];
    const float* Wv    = (const float*)d_in[3];   // [D, H*D]
    const float* Wo    = (const float*)d_in[4];   // [H*D, D]
    const float* bo    = (const float*)d_in[5];
    const float* gamma = (const float*)d_in[6];
    const float* beta  = (const float*)d_in[7];
    float* out = (float*)d_out;

    const size_t M   = (size_t)BATCH * SEQ;            // 8192 token rows
    const size_t nX  = M * DMODEL;                     // 4,194,304
    const size_t nWq = (size_t)DMODEL * HEADS * HK;    // 262,144
    const size_t nWv = (size_t)DMODEL * HEADS * HD;    // 2,097,152
    const size_t nWo = (size_t)HEADS * HD * DMODEL;    // 2,097,152
    const size_t nV  = M * HEADS * HD;                 // 33,554,432
    const size_t nAtt= (size_t)BATCH * SEQ * SEQ;      // 8,388,608 per head

    char* w = (char*)d_ws;
    __bf16* x_bf  = (__bf16*)w;              w += nX  * 2;
    __bf16* Wq_bf = (__bf16*)w;              w += nWq * 2;
    __bf16* Wk_bf = (__bf16*)w;              w += nWq * 2;
    __bf16* Wv_bf = (__bf16*)w;              w += nWv * 2;
    __bf16* Wo_bf = (__bf16*)w;              w += nWo * 2;
    __bf16* Q_bf  = (__bf16*)w;              w += nX  * 2;   // [B,S,H,HK]
    __bf16* K_bf  = (__bf16*)w;              w += nX  * 2;
    __bf16* V_bf  = (__bf16*)w;              w += nV  * 2;   // [B,S,H,HD]
    __bf16* O_bf  = (__bf16*)w;              w += nV  * 2;   // [B,S,H*HD]
    __bf16* att   = (__bf16*)w;              w += nAtt* 2;   // per-head, reused
    float*  proj  = (float*)w;               w += nX  * 4;

    const dim3 blk(256);
    auto cvt = [&](const float* src, __bf16* dst, size_t n) {
        int blocks = (int)((n + 255) / 256);
        if (blocks > 8192) blocks = 8192;
        cvt_bf16_kernel<<<blocks, blk, 0, stream>>>(src, dst, (int)n);
    };
    cvt(x,  x_bf,  nX);
    cvt(Wq, Wq_bf, nWq);
    cvt(Wk, Wk_bf, nWq);
    cvt(Wv, Wv_bf, nWv);
    cvt(Wo, Wo_bf, nWo);

    // Q = x @ Wq   (8192 x 512 x 512), bf16 out
    gemm_bf16_kernel<<<dim3(512/128, M/128, 1), blk, 0, stream>>>(
        x_bf, Wq_bf, Q_bf, (int)M, 512, 512, 512, 512, 512, 0, 0, 0, 1);
    // K = x @ Wk
    gemm_bf16_kernel<<<dim3(512/128, M/128, 1), blk, 0, stream>>>(
        x_bf, Wk_bf, K_bf, (int)M, 512, 512, 512, 512, 512, 0, 0, 0, 1);
    // V = x @ Wv   (8192 x 4096 x 512)
    gemm_bf16_kernel<<<dim3(4096/128, M/128, 1), blk, 0, stream>>>(
        x_bf, Wv_bf, V_bf, (int)M, 4096, 512, 512, 4096, 4096, 0, 0, 0, 1);

    // per head: scores+softmax -> att (L2-resident), then O_h = att @ V_h
    for (int h = 0; h < HEADS; ++h) {
        attn_softmax_kernel<<<dim3(SEQ/128, BATCH), blk, 0, stream>>>(
            Q_bf, K_bf, att, h);
        gemm_bf16_kernel<<<dim3(512/128, SEQ/128, BATCH), blk, 0, stream>>>(
            att, V_bf + (size_t)h * HD, O_bf + (size_t)h * HD,
            SEQ, HD, SEQ,
            /*lda*/SEQ, /*ldb*/HEADS * HD, /*ldc*/HEADS * HD,
            /*sAz*/(long long)SEQ * SEQ,
            /*sBz*/(long long)SEQ * HEADS * HD,
            /*sCz*/(long long)SEQ * HEADS * HD, 1);
    }

    // proj = O @ Wo   (8192 x 512 x 4096), f32 out
    gemm_bf16_kernel<<<dim3(512/128, M/128, 1), blk, 0, stream>>>(
        O_bf, Wo_bf, proj, (int)M, 512, 4096, 4096, 512, 512, 0, 0, 0, 0);

    // y = LN(proj + bo + x) * gamma + beta
    resid_ln_kernel<<<dim3((unsigned)M), blk, 0, stream>>>(proj, x, bo, gamma, beta, out);
}
